// SimGraphConstruction_46961172414952
// MI455X (gfx1250) — compile-verified, matrix operand order checked by reading, not run on hardware
//
#include <hip/hip_runtime.h>
#include <hip/hip_bf16.h>

// ---------------------------------------------------------------------------
// SimGraphConstruction for MI455X (gfx1250, wave32, WMMA)
//   N = 16384 rows, D = 256 features, K = 32 neighbors
//   out = [rows (K*N) | cols (K*N) | weights (K*N)] as float
//
// 32 rows per block (two 16-row M-tiles share one B stream -> half the L2
// traffic), fused running top-32 per row held one-slot-per-lane in a wave32.
// ---------------------------------------------------------------------------

#define DFEAT 256
#define KTOP  32
#define ROWS  32    // rows per block (two WMMA M-tiles)
#define CHUNK 256   // columns buffered in LDS per selection pass (32*256*4 = 32 KB)

typedef __bf16 bf16_t;
typedef bf16_t v8bf  __attribute__((ext_vector_type(8)));
typedef bf16_t v16bf __attribute__((ext_vector_type(16)));
typedef float  v8f   __attribute__((ext_vector_type(8)));

// ---------------- wave32 helpers ----------------

__device__ __forceinline__ float wave_min32(float v) {
    #pragma unroll
    for (int off = 16; off > 0; off >>= 1)
        v = fminf(v, __shfl_xor(v, off, 32));
    return v;
}

// Streaming top-K insert with cached wave-minimum. Fast path is cmp + ballot.
__device__ __forceinline__ void topk_insert(float &slotV, int &slotI, float &curMin,
                                            float candV, int candI, int lane) {
    unsigned long long mask = __ballot(candV > curMin);
    while (mask != 0ull) {
        const int src = __ffsll(mask) - 1;
        const float v  = __shfl(candV, src, 32);
        const int   ci = __shfl(candI, src, 32);
        if (v > curMin) {
            unsigned long long mm = __ballot(slotV == curMin);  // lane holding min
            const int dst = __ffsll(mm) - 1;
            if (lane == dst) { slotV = v; slotI = ci; }
            curMin = wave_min32(slotV);                         // refresh cache
        }
        if (lane == src) candV = -__builtin_inff();             // consumed
        mask = __ballot(candV > curMin);
    }
}

// Bitonic sort across 32 lanes, descending by value (lane i -> rank i).
__device__ __forceinline__ void sort32_desc(float &v, int &idx, int lane) {
    #pragma unroll
    for (int k = 2; k <= 32; k <<= 1) {
        #pragma unroll
        for (int j = k >> 1; j > 0; j >>= 1) {
            float ov = __shfl_xor(v, j, 32);
            int   oi = __shfl_xor(idx, j, 32);
            bool lower    = (lane & j) == 0;
            bool ascBlock = (lane & k) != 0;      // inverted -> final order descending
            bool keepMin  = (lower == ascBlock);
            bool takeOther = keepMin ? (ov < v) : (ov > v);
            if (takeOther) { v = ov; idx = oi; }
        }
    }
}

__device__ __forceinline__ v16bf make_frag(const v8bf lo, const v8bf hi) {
    return __builtin_shufflevector(lo, hi,
             0,1,2,3,4,5,6,7,8,9,10,11,12,13,14,15);
}

// ---------------- kernel 1: row L2-normalize, fp32 -> bf16 ----------------

__global__ __launch_bounds__(256)
void normalize_bf16_kernel(const float* __restrict__ feat,
                           bf16_t* __restrict__ nrm, int N) {
    __shared__ float red[8];
    const int r = blockIdx.x;
    const int t = threadIdx.x;          // 256 threads == DFEAT
    float v = feat[(size_t)r * DFEAT + t];
    float s = v * v;
    #pragma unroll
    for (int off = 16; off > 0; off >>= 1) s += __shfl_xor(s, off, 32);
    if ((t & 31) == 0) red[t >> 5] = s;
    __syncthreads();
    float tot = 0.f;
    #pragma unroll
    for (int i = 0; i < 8; ++i) tot += red[i];
    const float scale = 1.0f / (sqrtf(tot) + 1e-10f);
    nrm[(size_t)r * DFEAT + t] = (bf16_t)(v * scale);
}

// ---------------- kernel 2: fused WMMA GEMM + running top-K ----------------

__global__ __launch_bounds__(256)
void sim_topk_kernel(const bf16_t* __restrict__ nrm,
                     float* __restrict__ out, int N) {
    __shared__ __align__(16) bf16_t ldsA[ROWS][DFEAT];     // 16 KB: block's rows
    __shared__ __align__(16) float  simb[ROWS][CHUNK];     // 32 KB: sim chunk

    const int tid  = threadIdx.x;
    const int lane = tid & 31;
    const int wave = tid >> 5;
    const int rowBase = blockIdx.x * ROWS;

    // ---- stage the 32 A-rows through LDS (coalesced 16B copies) ----
    {
        const uint4* g = (const uint4*)(nrm + (size_t)rowBase * DFEAT);
        uint4* s = (uint4*)(&ldsA[0][0]);
        const int nvec = ROWS * DFEAT * (int)sizeof(bf16_t) / 16;  // 1024
        for (int i = tid; i < nvec; i += 256) s[i] = g[i];
    }
    __syncthreads();

    // ---- A fragments (16x32 bf16 WMMA layout) ----
    // lanes 0-15: row M=lane,   K = {k0..k0+7, k0+16..k0+23}
    // lanes16-31: row M=lane-16,K = {k0+8..k0+15, k0+24..k0+31}
    const int arow  = lane & 15;
    const int khalf = (lane >> 4) * 8;

    // Low M-tile (rows rowBase+0..15): persistent in VGPRs (reused ~2048x).
    v16bf afragLo[8];
    #pragma unroll
    for (int kk = 0; kk < 8; ++kk) {
        const int k0 = kk * 32;
        afragLo[kk] = make_frag(*(const v8bf*)&ldsA[arow][k0 + khalf],
                                *(const v8bf*)&ldsA[arow][k0 + 16 + khalf]);
    }

    // ---- per-wave running top-32 state (one slot per lane), 4 rows/wave ----
    const int r0 = wave * 4;
    float topV0 = -__builtin_inff(), topV1 = topV0, topV2 = topV0, topV3 = topV0;
    float curMin0 = topV0, curMin1 = topV0, curMin2 = topV0, curMin3 = topV0;
    int   topI0 = 0, topI1 = 0, topI2 = 0, topI3 = 0;

    for (int c0 = 0; c0 < N; c0 += CHUNK) {
        // ===== GEMM phase: 8 waves x 2 tiles cover CHUNK=256 columns =====
        #pragma unroll
        for (int t = 0; t < CHUNK / (8 * 16); ++t) {
            const int colTile = c0 + (t * 8 + wave) * 16;
            const int bcol    = colTile + (lane & 15);
            const bf16_t* brow = nrm + (size_t)bcol * DFEAT;

            // prefetch next tile's B row into L2 (speculative, may drop)
            {
                const int pcol = bcol + 8 * 16;
                if (pcol < N)
                    __builtin_prefetch(nrm + (size_t)pcol * DFEAT, 0, 1);
            }

            // Batch-load ALL 8 B fragments (16x global_load_b128), then pin
            // them ahead of the WMMA chain with a scheduling barrier so the
            // backend cannot re-serialize into load->wait->wmma. Waits can
            // then be staged (loadcnt<=14,<=12,...) overlapping L2 latency.
            v16bf bfr[8];
            #pragma unroll
            for (int kk = 0; kk < 8; ++kk) {
                const int k0 = kk * 32;
                bfr[kk] = make_frag(*(const v8bf*)(brow + k0 + khalf),
                                    *(const v8bf*)(brow + k0 + 16 + khalf));
            }
            asm volatile("" ::: "memory");   // keep the load batch ahead

            // Two interleaved accumulator chains: low tile from registers,
            // high tile A-fragments streamed from LDS (ds latency hides
            // under the low-chain WMMAs). Each B fragment feeds 2 WMMAs.
            v8f accLo = {0.f,0.f,0.f,0.f,0.f,0.f,0.f,0.f};
            v8f accHi = {0.f,0.f,0.f,0.f,0.f,0.f,0.f,0.f};
            #pragma unroll
            for (int kk = 0; kk < 8; ++kk) {
                const int k0 = kk * 32;
                v16bf aHi = make_frag(
                    *(const v8bf*)&ldsA[16 + arow][k0 + khalf],
                    *(const v8bf*)&ldsA[16 + arow][k0 + 16 + khalf]);
                accLo = __builtin_amdgcn_wmma_f32_16x16x32_bf16(
                            false, afragLo[kk], false, bfr[kk],
                            (short)0, accLo, false, false);
                accHi = __builtin_amdgcn_wmma_f32_16x16x32_bf16(
                            false, aHi, false, bfr[kk],
                            (short)0, accHi, false, false);
            }

            // scatter to LDS; C layout: VGPR j -> M = j + 8*(lane>=16),
            // N = lane&15. Zero the diagonal (reference: sim - diag(sim)).
            const int lcol  = colTile - c0 + (lane & 15);
            const int mbase = (lane >> 4) * 8;
            #pragma unroll
            for (int j = 0; j < 8; ++j) {
                float vLo = accLo[j];
                float vHi = accHi[j];
                if (rowBase + mbase + j == bcol)      vLo = 0.0f;
                if (rowBase + 16 + mbase + j == bcol) vHi = 0.0f;
                simb[mbase + j][lcol]      = vLo;
                simb[16 + mbase + j][lcol] = vHi;
            }
        }
        __syncthreads();

        // ===== selection phase: each wave scans 4 rows of the chunk =====
        for (int j = lane; j < CHUNK; j += 32) {
            const int c = c0 + j;
            topk_insert(topV0, topI0, curMin0, simb[r0 + 0][j], c, lane);
            topk_insert(topV1, topI1, curMin1, simb[r0 + 1][j], c, lane);
            topk_insert(topV2, topI2, curMin2, simb[r0 + 2][j], c, lane);
            topk_insert(topV3, topI3, curMin3, simb[r0 + 3][j], c, lane);
        }
        __syncthreads();
    }

    // ---- final: sort descending, emit edge/weights in rank-major order ----
    sort32_desc(topV0, topI0, lane);
    sort32_desc(topV1, topI1, lane);
    sort32_desc(topV2, topI2, lane);
    sort32_desc(topV3, topI3, lane);

    const size_t KN = (size_t)KTOP * (size_t)N;
    const int n0 = rowBase + r0, n1 = n0 + 1, n2 = n0 + 2, n3 = n0 + 3;
    // edge row 0 (source node ids), edge row 1 (neighbor ids), weights
    out[(size_t)lane * N + n0]          = (float)n0;
    out[(size_t)lane * N + n1]          = (float)n1;
    out[(size_t)lane * N + n2]          = (float)n2;
    out[(size_t)lane * N + n3]          = (float)n3;
    out[KN + (size_t)lane * N + n0]     = (float)topI0;
    out[KN + (size_t)lane * N + n1]     = (float)topI1;
    out[KN + (size_t)lane * N + n2]     = (float)topI2;
    out[KN + (size_t)lane * N + n3]     = (float)topI3;
    out[2 * KN + (size_t)lane * N + n0] = topV0;
    out[2 * KN + (size_t)lane * N + n1] = topV1;
    out[2 * KN + (size_t)lane * N + n2] = topV2;
    out[2 * KN + (size_t)lane * N + n3] = topV3;
}

// ---------------------------------------------------------------------------

extern "C" void kernel_launch(void* const* d_in, const int* in_sizes, int n_in,
                              void* d_out, int out_size, void* d_ws, size_t ws_size,
                              hipStream_t stream) {
    (void)n_in; (void)out_size; (void)ws_size;
    const float* feat = (const float*)d_in[0];
    float* out = (float*)d_out;
    bf16_t* nrm = (bf16_t*)d_ws;                 // N*D bf16 = 8 MB scratch
    const int N = in_sizes[0] / DFEAT;           // 16384

    normalize_bf16_kernel<<<N, 256, 0, stream>>>(feat, nrm, N);
    sim_topk_kernel<<<N / ROWS, 256, 0, stream>>>(nrm, out, N);
}
